// TaggingCRF_43387759624606
// MI455X (gfx1250) — compile-verified, matrix operand order checked by reading, not run on hardware
//
#include <hip/hip_runtime.h>
#include <hip/hip_bf16.h>

// CRF (B=256, T=1024, K=128) for MI455X / gfx1250, wave32.
// Forward logsumexp recursion is executed on the matrix cores by rewriting
// logsumexp_i(s_i + T_ij) = m + log( sum_i exp(s_i - m) * exp(T_ij) )
// i.e. a f16 GEMM (p  x  expT) with f32 accumulation via
// v_wmma_f32_16x16x32_f16.  Viterbi (max-plus) is VALU with the 64 KB
// transition matrix DMA'd into LDS by the Tensor Data Mover.
//
// d_ws layout:
//   [0, 256*1023*128)                  uint8 backpointers  (~32.0 MB)
//   [32MB + 0*1024)                    float num[256]
//   [32MB + 1*1024)                    float den[256]
//   [32MB + 2*1024)                    int   best_last[256]

#define Bc 256
#define Tc 1024
#define Kc 128

typedef __attribute__((ext_vector_type(16))) _Float16 v16h;
typedef __attribute__((ext_vector_type(8)))  float    v8f;
typedef __attribute__((ext_vector_type(4)))  unsigned int u32x4;
typedef __attribute__((ext_vector_type(4)))  int          i32x4;
typedef __attribute__((ext_vector_type(8)))  int          i32x8;

#if defined(__has_builtin)
#if __has_builtin(__builtin_amdgcn_tensor_load_to_lds) && \
    __has_builtin(__builtin_amdgcn_s_wait_tensorcnt)
#define HAVE_TDM 1
#endif
#endif

// ---------------------------------------------------------------------------
// Kernel 1: numerator (gold path score) per batch. One block per batch,
// deterministic LDS tree reduction (no float atomics).
// ---------------------------------------------------------------------------
__global__ __launch_bounds__(256) void crf_num_kernel(
    const float* __restrict__ logits, const int* __restrict__ labels,
    const int* __restrict__ mask, const float* __restrict__ start_t,
    const float* __restrict__ end_t, const float* __restrict__ trans,
    float* __restrict__ num)
{
    __shared__ float redS[256];
    __shared__ int   redM[256];
    const int b = blockIdx.x, tid = threadIdx.x;
    float acc = 0.f; int mcnt = 0;
    for (int t = tid; t < Tc; t += 256) {
        const int tag = labels[b * Tc + t];
        const int mv  = mask[b * Tc + t];
        mcnt += mv;
        if (t == 0) {
            acc += start_t[tag] + logits[(size_t)b * Tc * Kc + tag];
        } else if (mv) {
            const int ptag = labels[b * Tc + t - 1];
            acc += trans[ptag * Kc + tag] +
                   logits[((size_t)b * Tc + t) * Kc + tag];
        }
    }
    redS[tid] = acc; redM[tid] = mcnt;
    __syncthreads();
    for (int s = 128; s > 0; s >>= 1) {
        if (tid < s) { redS[tid] += redS[tid + s]; redM[tid] += redM[tid + s]; }
        __syncthreads();
    }
    if (tid == 0) {
        const int send = redM[0] - 1;                 // seq_ends = sum(mask)-1
        num[b] = redS[0] + end_t[labels[b * Tc + send]];
    }
}

// ---------------------------------------------------------------------------
// Kernel 2: forward logsumexp recursion via WMMA.  16 blocks x 256 threads.
// Block handles batches [16*blk, 16*blk+16); wave w owns output columns
// [16w, 16w+16).  expT B-tiles live in VGPRs for the whole T loop.
// ---------------------------------------------------------------------------
__global__ __launch_bounds__(256) void crf_forward_lse_kernel(
    const float* __restrict__ logits, const int* __restrict__ mask,
    const float* __restrict__ start_t, const float* __restrict__ end_t,
    const float* __restrict__ trans, float* __restrict__ den)
{
    __shared__ float    scoreS[16 * Kc];   // running log-scores, 8 KB
    __shared__ _Float16 pS[16 * Kc];       // exp(score - m), 4 KB
    __shared__ float    mS[16];            // per-batch row max
    __shared__ float    red[16 * 16];

    const int tid  = threadIdx.x;
    const int wave = tid >> 5;             // 0..7  -> j tile
    const int lane = tid & 31;
    const int b0   = blockIdx.x * 16;
    const int mrow = lane & 15;            // M for A-tile, N for B/D tiles
    const int g    = lane >> 4;            // lane half-group
    const int j0   = wave * 16;

    // Persistent B tiles: expT[k][j0+n], 4 chunks of K=32.
    // 16-bit B layout: lane -> N = lane&15; half h -> K = (h/8)*16 + (h&7) + 8g.
    v16h btile[4];
    #pragma unroll
    for (int c = 0; c < 4; ++c) {
        #pragma unroll
        for (int h = 0; h < 16; ++h) {
            const int kk = c * 32 + (h >> 3) * 16 + (h & 7) + 8 * g;
            btile[c][h] = (_Float16)__expf(trans[kk * Kc + j0 + mrow]);
        }
    }

    // score0 = start + emis[0]
    for (int idx = tid; idx < 16 * Kc; idx += 256) {
        const int bb = idx >> 7, jj = idx & 127;
        scoreS[idx] = start_t[jj] + logits[(size_t)(b0 + bb) * Tc * Kc + jj];
    }
    __syncthreads();

    for (int t = 1; t < Tc; ++t) {
        // --- 1) per-row max m_b (2048 elems, 8 per thread + 16-way finish)
        {
            const int bb = tid >> 4, seg = tid & 15;
            const float* r = &scoreS[bb * Kc + seg * 8];
            float mx = r[0];
            #pragma unroll
            for (int q = 1; q < 8; ++q) mx = fmaxf(mx, r[q]);
            red[bb * 16 + seg] = mx;
        }
        __syncthreads();
        if (tid < 16) {
            float mx = red[tid * 16];
            #pragma unroll
            for (int q = 1; q < 16; ++q) mx = fmaxf(mx, red[tid * 16 + q]);
            mS[tid] = mx;
        }
        __syncthreads();
        // --- 2) p = exp(score - m) -> f16 in LDS
        {
            const int bb = tid >> 4, seg = tid & 15;
            const float mb = mS[bb];
            #pragma unroll
            for (int q = 0; q < 8; ++q) {
                const int jj = seg * 8 + q;
                pS[bb * Kc + jj] =
                    (_Float16)__expf(scoreS[bb * Kc + jj] - mb);
            }
        }
        __syncthreads();
        // --- 3) A-tile loads (2 x ds_load_b128 per tile) + 4 chained WMMAs
        v8f acc = {};
        #pragma unroll
        for (int c = 0; c < 4; ++c) {
            v16h a;
            const _Float16* lo = &pS[mrow * Kc + c * 32 + 8 * g]; // K = 8g..8g+7
            const _Float16* hi = lo + 16;                          // K += 16
            #pragma unroll
            for (int h = 0; h < 8; ++h) { a[h] = lo[h]; a[h + 8] = hi[h]; }
            acc = __builtin_amdgcn_wmma_f32_16x16x32_f16(
                false, a, false, btile[c], (short)0, acc, false, false);
        }
        // --- 4) epilogue: score' = m + log(acc) + emis (masked)
        #pragma unroll
        for (int r = 0; r < 8; ++r) {
            const int M  = r + 8 * g;          // D layout: M = vgpr + 8*(lane/16)
            const int bb = b0 + M;
            const int jj = j0 + mrow;
            const size_t eidx = ((size_t)bb * Tc + t) * Kc + jj;
            const float emis = logits[eidx];
            if (t + 1 < Tc) __builtin_prefetch(&logits[eidx + Kc], 0, 1);
            const float old = scoreS[M * Kc + jj];
            const float nw  = mS[M] + __logf(acc[r]) + emis;
            scoreS[M * Kc + jj] = mask[bb * Tc + t] ? nw : old;
        }
        __syncthreads();
    }

    // --- den_b = logsumexp_j(score + end)
    {
        const int bb = tid >> 4, seg = tid & 15;
        float mx = -3.4e38f;
        #pragma unroll
        for (int q = 0; q < 8; ++q) {
            const int jj = seg * 8 + q;
            mx = fmaxf(mx, scoreS[bb * Kc + jj] + end_t[jj]);
        }
        red[bb * 16 + seg] = mx;
    }
    __syncthreads();
    if (tid < 16) {
        float mx = red[tid * 16];
        #pragma unroll
        for (int q = 1; q < 16; ++q) mx = fmaxf(mx, red[tid * 16 + q]);
        mS[tid] = mx;
    }
    __syncthreads();
    {
        const int bb = tid >> 4, seg = tid & 15;
        float s = 0.f;
        #pragma unroll
        for (int q = 0; q < 8; ++q) {
            const int jj = seg * 8 + q;
            s += __expf(scoreS[bb * Kc + jj] + end_t[jj] - mS[bb]);
        }
        red[bb * 16 + seg] = s;
    }
    __syncthreads();
    if (tid < 16) {
        float s = 0.f;
        #pragma unroll
        for (int q = 0; q < 16; ++q) s += red[tid * 16 + q];
        den[b0 + tid] = mS[tid] + __logf(s);
    }
}

// ---------------------------------------------------------------------------
// Kernel 3: Viterbi forward (max-plus).  2 batches per block; full 64 KB
// trans matrix resident in LDS (CDNA5: up to 320 KB per workgroup), loaded
// by the Tensor Data Mover as one 128x128 f32 2-D tile (TENSORcnt-tracked).
// ---------------------------------------------------------------------------
__global__ __launch_bounds__(256) void crf_viterbi_kernel(
    const float* __restrict__ logits, const int* __restrict__ mask,
    const float* __restrict__ start_t, const float* __restrict__ end_t,
    const float* __restrict__ trans,
    unsigned char* __restrict__ hist, int* __restrict__ best_last)
{
    __shared__ float transS[Kc * Kc];      // 64 KB
    __shared__ float scoreS[2 * Kc];
    __shared__ float redv[2 * Kc];
    __shared__ int   redi[2 * Kc];

    const int tid = threadIdx.x;
    const int lb  = tid >> 7;              // local batch 0/1
    const int j   = tid & 127;
    const int b   = blockIdx.x * 2 + lb;

#if defined(HAVE_TDM)
    if (tid < 32) {                        // one wave issues the DMA
        // D# group 0: count=1 | lds_addr | 57-bit global addr | type=2
        const unsigned lds_off = (unsigned)(unsigned long long)(size_t)transS;
        const unsigned long long ga = (unsigned long long)(size_t)trans;
        u32x4 g0 = { 1u, lds_off, (unsigned)ga,
                     ((unsigned)((ga >> 32) & 0x01FFFFFFull)) | (2u << 30) };
        // D# group 1: wg_mask=0, data_size=4B; tensor_dim0/1=128,
        // tile_dim0/1=128, tensor_dim0_stride=128 (units of data_size)
        i32x8 g1 = { 0x00020000, (int)(128u << 16), (int)(128u << 16),
                     (int)(128u << 16), 128, 128, 0, 0 };
        i32x4 z4 = { 0, 0, 0, 0 };
#if __clang_major__ >= 23
        i32x8 z8 = { 0, 0, 0, 0, 0, 0, 0, 0 };
        __builtin_amdgcn_tensor_load_to_lds(g0, g1, z4, z4, z8, 0);
#else
        __builtin_amdgcn_tensor_load_to_lds(g0, g1, z4, z4, 0);
#endif
        __builtin_amdgcn_s_wait_tensorcnt(0);  // drain TENSORcnt before publish
    }
#else
    for (int idx = tid; idx < Kc * Kc; idx += 256) transS[idx] = trans[idx];
#endif
    scoreS[lb * Kc + j] = start_t[j] + logits[(size_t)b * Tc * Kc + j];
    __syncthreads();

    for (int t = 1; t < Tc; ++t) {
        const float* sc = &scoreS[lb * Kc];
        float best = -3.4e38f; int bi = 0;
        #pragma unroll 4
        for (int i = 0; i < Kc; ++i) {
            const float cand = sc[i] + transS[i * Kc + j];
            if (cand > best) { best = cand; bi = i; }   // first-max tie-break
        }
        const size_t eidx = ((size_t)b * Tc + t) * Kc + j;
        const float emis = logits[eidx];
        if (t + 1 < Tc) __builtin_prefetch(&logits[eidx + Kc], 0, 1);
        const int   mv   = mask[b * Tc + t];
        hist[((size_t)b * (Tc - 1) + (t - 1)) * Kc + j] = (unsigned char)bi;
        const float old = sc[j];
        __syncthreads();
        scoreS[lb * Kc + j] = mv ? (best + emis) : old;
        __syncthreads();
    }

    // best_last = argmax_j(score + end), first-max semantics
    redv[lb * Kc + j] = scoreS[lb * Kc + j] + end_t[j];
    redi[lb * Kc + j] = j;
    __syncthreads();
    for (int s = 64; s > 0; s >>= 1) {
        if (j < s) {
            if (redv[lb * Kc + j + s] > redv[lb * Kc + j]) {
                redv[lb * Kc + j] = redv[lb * Kc + j + s];
                redi[lb * Kc + j] = redi[lb * Kc + j + s];
            }
        }
        __syncthreads();
    }
    if (j == 0) best_last[b] = redi[lb * Kc];
}

// ---------------------------------------------------------------------------
// Kernel 4: backtrace.  One thread per batch; history is L2-resident (33.5 MB
// < 192 MB global L2).  Writes tags as float into d_out[0 .. B*T).
// ---------------------------------------------------------------------------
__global__ void crf_backtrace_kernel(
    const unsigned char* __restrict__ hist, const int* __restrict__ mask,
    const int* __restrict__ best_last, float* __restrict__ out_tags)
{
    const int b = blockIdx.x * blockDim.x + threadIdx.x;
    if (b >= Bc) return;
    int carry = best_last[b];
    for (int t = Tc - 1; t >= 1; --t) {
        const int mv = mask[b * Tc + t];
        out_tags[b * Tc + t] = (float)(mv ? carry : 0);
        const int prev = hist[((size_t)b * (Tc - 1) + (t - 1)) * Kc + carry];
        carry = mv ? prev : carry;
    }
    out_tags[b * Tc] = (float)carry;
}

// ---------------------------------------------------------------------------
// Kernel 5: nll = sum_b (den_b - num_b)  ->  d_out[B*T]
// ---------------------------------------------------------------------------
__global__ __launch_bounds__(256) void crf_finalize_kernel(
    const float* __restrict__ num, const float* __restrict__ den,
    float* __restrict__ out_nll)
{
    __shared__ float red[256];
    const int tid = threadIdx.x;
    red[tid] = den[tid] - num[tid];
    __syncthreads();
    for (int s = 128; s > 0; s >>= 1) {
        if (tid < s) red[tid] += red[tid + s];
        __syncthreads();
    }
    if (tid == 0) out_nll[0] = red[0];
}

extern "C" void kernel_launch(void* const* d_in, const int* in_sizes, int n_in,
                              void* d_out, int out_size, void* d_ws, size_t ws_size,
                              hipStream_t stream) {
    const float* logits  = (const float*)d_in[0];
    const int*   labels  = (const int*)d_in[1];
    const int*   mask    = (const int*)d_in[2];
    const float* start_t = (const float*)d_in[3];
    const float* end_t   = (const float*)d_in[4];
    const float* trans   = (const float*)d_in[5];
    float* out = (float*)d_out;               // [B*T tags] ++ [nll]

    unsigned char* hist = (unsigned char*)d_ws;               // 32.0 MB
    char* base = (char*)d_ws + ((size_t)32 << 20);
    float* num       = (float*)(base);
    float* den       = (float*)(base + 1024);
    int*   best_last = (int*)  (base + 2048);

    crf_num_kernel<<<Bc, 256, 0, stream>>>(logits, labels, mask, start_t,
                                           end_t, trans, num);
    crf_forward_lse_kernel<<<Bc / 16, 256, 0, stream>>>(logits, mask, start_t,
                                                        end_t, trans, den);
    crf_viterbi_kernel<<<Bc / 2, 256, 0, stream>>>(logits, mask, start_t,
                                                   end_t, trans, hist, best_last);
    crf_backtrace_kernel<<<(Bc + 63) / 64, 64, 0, stream>>>(hist, mask,
                                                            best_last, out);
    crf_finalize_kernel<<<1, 256, 0, stream>>>(num, den, out + (size_t)Bc * Tc);
}
